// MultiTaskDecoders_40561671143595
// MI455X (gfx1250) — matrix operand order repaired and checked by reading, not compiled, not run on hardware
//
#include <hip/hip_runtime.h>

typedef __bf16 bf16_t;
typedef __attribute__((ext_vector_type(16))) __bf16 v16bf;
typedef __attribute__((ext_vector_type(8)))  __bf16 v8bf;
typedef __attribute__((ext_vector_type(4)))  __bf16 v4bf;
typedef __attribute__((ext_vector_type(8)))  float  v8f;
typedef __attribute__((ext_vector_type(4)))  float  v4f;

#define B_ROWS   131072
#define KTOT     256      // input dim
#define NTOT     768      // concat hidden: 128+128+256+256
#define MTILE    64       // rows per workgroup
#define NTILES   48       // NTOT/16
#define KSTEPS   8        // KTOT/32

// ---------------------------------------------------------------------------
// Pack W1_{sc,st,women,children} (fp32, [256 x h] row-major) into bf16
// B-fragments for V_WMMA_F32_16X16X32_BF16 in contiguous k order (A uses the same) per lane:
//   fragment (ntile, kstep): lane l holds column n = ntile*16 + (l&15),
//   elements e=0..15 hold K = kstep*32 + (l>>4)*16 + e.
// Stored contiguously: wpack[((ntile*8 + k)*32 + lane)*16 + e]
// ---------------------------------------------------------------------------
__global__ void pack_w1(const float* __restrict__ w1sc, const float* __restrict__ w1st,
                        const float* __restrict__ w1w,  const float* __restrict__ w1c,
                        bf16_t* __restrict__ wpack) {
  const int nt = blockIdx.x;       // 0..47
  const int t  = threadIdx.x;      // 0..255
  const int lane = t & 31;
  const int k    = t >> 5;         // 0..7
  const int n    = nt * 16 + (lane & 15);
  const int half = lane >> 4;
  bf16_t* dst = wpack + (((nt * KSTEPS + k) * 32 + lane) * 16);
#pragma unroll
  for (int e = 0; e < 16; ++e) {
    const int kk = k * 32 + half * 16 + e;
    float v;
    if      (n < 128) v = w1sc[kk * 128 + n];
    else if (n < 256) v = w1st[kk * 128 + (n - 128)];
    else if (n < 512) v = w1w [kk * 256 + (n - 256)];
    else              v = w1c [kk * 256 + (n - 512)];
    dst[e] = (bf16_t)v;
  }
}

// Concatenate b1 / W2 / b2 across experts into fp32 vectors in workspace.
__global__ void pack_vec(const float* __restrict__ b1sc, const float* __restrict__ b1st,
                         const float* __restrict__ b1w,  const float* __restrict__ b1c,
                         const float* __restrict__ w2sc, const float* __restrict__ w2st,
                         const float* __restrict__ w2w,  const float* __restrict__ w2c,
                         const float* __restrict__ b2sc, const float* __restrict__ b2st,
                         const float* __restrict__ b2w,  const float* __restrict__ b2c,
                         float* __restrict__ b1, float* __restrict__ w2,
                         float* __restrict__ b2) {
  const int i = blockIdx.x * blockDim.x + threadIdx.x;
  if (i < 768) {
    float bv, wv;
    if      (i < 128) { bv = b1sc[i];       wv = w2sc[i];       }
    else if (i < 256) { bv = b1st[i - 128]; wv = w2st[i - 128]; }
    else if (i < 512) { bv = b1w [i - 256]; wv = w2w [i - 256]; }
    else              { bv = b1c [i - 512]; wv = w2c [i - 512]; }
    b1[i] = bv; w2[i] = wv;
  }
  if (i == 0) { b2[0] = b2sc[0]; b2[1] = b2st[0]; b2[2] = b2w[0]; b2[3] = b2c[0]; }
}

// ---------------------------------------------------------------------------
// Fused routed-MoE forward: H = relu(x @ W1all + b1); y = H_e . W2_e + b2_e
// One block = 64 rows x all 768 hidden cols; 16 waves; each wave:
//   row tile rt = wid&3, column group cg = wid>>2 (12 of 48 n-tiles).
// ---------------------------------------------------------------------------
__global__ __launch_bounds__(512)
void expert_gemm(const float* __restrict__ x, const int* __restrict__ labels,
                 const bf16_t* __restrict__ wpack, const float* __restrict__ b1,
                 const float* __restrict__ w2, const float* __restrict__ b2,
                 float* __restrict__ out) {
  __shared__ __align__(16) bf16_t xs[MTILE][KTOT];  // 32 KB bf16 x-tile
  __shared__ float part[MTILE][4];                  // per-(row, expert) partials

  const int tid = threadIdx.x;
  const size_t rowbase = (size_t)blockIdx.x * MTILE;

  // zero per-row/expert partials
  if (tid < MTILE * 4) ((float*)part)[tid] = 0.0f;

  // Stage x tile -> LDS as bf16. 64x256 fp32 = 16384 floats; 512 thr x 8 iters
  // x 4 floats, fully coalesced b128 loads, b64 LDS stores.
  {
    const float* src = x + rowbase * KTOT;
#pragma unroll
    for (int i = 0; i < 8; ++i) {
      const int flat = i * 2048 + tid * 4;
      const int r = flat >> 8;
      const int c = flat & 255;
      v4f v = *(const v4f*)(src + flat);
      v4bf o;
      o[0] = (bf16_t)v[0]; o[1] = (bf16_t)v[1];
      o[2] = (bf16_t)v[2]; o[3] = (bf16_t)v[3];
      *(v4bf*)(&xs[r][c]) = o;
    }
  }
  __syncthreads();

  const int lane = tid & 31;
  const int wid  = tid >> 5;      // 0..15
  const int rt   = wid & 3;       // row tile within block
  const int cg   = wid >> 2;      // column group (12 n-tiles each)
  const int l15  = lane & 15;
  const int half = lane >> 4;
  const int m    = rt * 16 + l15; // A-matrix row this lane feeds

  v8f acc[12];
  const v8f vzero = {0.f, 0.f, 0.f, 0.f, 0.f, 0.f, 0.f, 0.f};
#pragma unroll
  for (int t = 0; t < 12; ++t) acc[t] = vzero;

#pragma unroll
  for (int k = 0; k < KSTEPS; ++k) {
    // A fragment (matches B packing): e -> K = k*32 + half*16 + e
    v16bf afrag;
    {
      const bf16_t* base = &xs[m][k * 32 + half * 16];
      v8bf c0 = *(const v8bf*)base;
      v8bf c1 = *(const v8bf*)(base + 8);
#pragma unroll
      for (int i = 0; i < 8; ++i) { afrag[i] = c0[i]; afrag[i + 8] = c1[i]; }
    }
#pragma unroll
    for (int t = 0; t < 12; ++t) {
      const int nt = cg * 12 + t;
      const bf16_t* bp = wpack + (((nt * KSTEPS + k) * 32 + lane) * 16);
      v16bf bfrag = *(const v16bf*)bp;  // 32 B/lane, pre-packed
      acc[t] = __builtin_amdgcn_wmma_f32_16x16x32_bf16(
          false, afrag, false, bfrag, (short)0, acc[t], false, false);
    }
  }

  // Fused epilogue: bias + ReLU + scale by W2 column weight, reduce the 16
  // columns of each n-tile across lanes, accumulate into (row, expert).
#pragma unroll
  for (int t = 0; t < 12; ++t) {
    const int nt = cg * 12 + t;
    const int n  = nt * 16 + l15;
    const int e  = (nt < 8) ? 0 : (nt < 16) ? 1 : (nt < 32) ? 2 : 3;
    const float cb = b1[n];
    const float cw = w2[n];
#pragma unroll
    for (int v = 0; v < 8; ++v) {
      float h = acc[t][v] + cb;
      h = h > 0.0f ? h : 0.0f;
      float s = h * cw;
      // reduce across the 16 lanes of this half (columns n0..n0+15)
      s += __shfl_xor(s, 1, 16);
      s += __shfl_xor(s, 2, 16);
      s += __shfl_xor(s, 4, 16);
      s += __shfl_xor(s, 8, 16);
      if (l15 == 0) {
        // C/D layout: VGPR v, lanes 0-15 -> M=v, lanes 16-31 -> M=v+8
        atomicAdd(&part[rt * 16 + half * 8 + v][e], s);
      }
    }
  }
  __syncthreads();

  // Route: each of the 64 rows picks its labeled expert.
  if (tid < MTILE) {
    const size_t row = rowbase + tid;
    const int lbl = labels[row];
    out[row] = part[tid][lbl] + b2[lbl];
  }
}

// ---------------------------------------------------------------------------
extern "C" void kernel_launch(void* const* d_in, const int* in_sizes, int n_in,
                              void* d_out, int out_size, void* d_ws, size_t ws_size,
                              hipStream_t stream) {
  const float* x      = (const float*)d_in[0];
  const int*   labels = (const int*)d_in[1];
  const float* W1sc = (const float*)d_in[2];
  const float* b1sc = (const float*)d_in[3];
  const float* W2sc = (const float*)d_in[4];
  const float* b2sc = (const float*)d_in[5];
  const float* W1st = (const float*)d_in[6];
  const float* b1st = (const float*)d_in[7];
  const float* W2st = (const float*)d_in[8];
  const float* b2st = (const float*)d_in[9];
  const float* W1w  = (const float*)d_in[10];
  const float* b1w  = (const float*)d_in[11];
  const float* W2w  = (const float*)d_in[12];
  const float* b2w  = (const float*)d_in[13];
  const float* W1c  = (const float*)d_in[14];
  const float* b1c  = (const float*)d_in[15];
  const float* W2c  = (const float*)d_in[16];
  const float* b2c  = (const float*)d_in[17];

  char* ws = (char*)d_ws;
  bf16_t* wpack = (bf16_t*)ws;                              // 48*8*32*16 bf16 = 384 KB
  float*  b1    = (float*)(ws + 393216);                    // 768 fp32
  float*  w2    = (float*)(ws + 393216 + 3072);             // 768 fp32
  float*  b2    = (float*)(ws + 393216 + 6144);             // 4 fp32

  pack_w1<<<NTILES, 256, 0, stream>>>(W1sc, W1st, W1w, W1c, wpack);
  pack_vec<<<2, 512, 0, stream>>>(b1sc, b1st, b1w, b1c,
                                  W2sc, W2st, W2w, W2c,
                                  b2sc, b2st, b2w, b2c,
                                  b1, w2, b2);
  expert_gemm<<<B_ROWS / MTILE, 512, 0, stream>>>(x, labels, wpack, b1, w2, b2,
                                                  (float*)d_out);
}
